// DSSMReverse_80006650790076
// MI455X (gfx1250) — matrix operand order repaired and checked by reading, not compiled, not run on hardware
//
#include <hip/hip_runtime.h>
#include <hip/hip_bf16.h>

typedef __bf16 bf16x16 __attribute__((ext_vector_type(16)));
typedef float v8f __attribute__((ext_vector_type(8)));

#define B_SZ   4096
#define SE_C   8
#define C1_C   64
#define C2_C   128
#define E_C    256
#define NZ_C   1024
#define K2_C   576      // C1*9
#define FEAT   8192     // C2*64

// ---------------------------------------------------------------------------
// Stage 0: embedding max-norm renorm (64 rows x 8)
// ---------------------------------------------------------------------------
__global__ void k_norm_embed(const float* __restrict__ w_embed, float* __restrict__ w_norm) {
    int r = threadIdx.x;           // 64 threads
    float v[8]; float s = 0.f;
    for (int c = 0; c < 8; ++c) { v[c] = w_embed[r * 8 + c]; s += v[c] * v[c]; }
    float n = sqrtf(s);
    float inv = fminf(1.f, 1.f / fmaxf(n, 1e-7f));
    for (int c = 0; c < 8; ++c) w_norm[r * 8 + c] = v[c] * inv;
}

// ---------------------------------------------------------------------------
// f32 -> bf16 convert
// ---------------------------------------------------------------------------
__global__ void k_cvt_bf16(const float* __restrict__ x, __bf16* __restrict__ y, int n) {
    int i = blockIdx.x * 256 + threadIdx.x;
    if (i < n) y[i] = (__bf16)x[i];
}

// ---------------------------------------------------------------------------
// Fused per-sample branch: gather -> conv1(+relu, VALU) -> im2col(LDS)
//   -> conv2 as WMMA GEMM [128 x 64] over K=576 -> bias+relu -> y2 bf16
// One workgroup (128 threads = 4 waves) per sample. K chunked 2 x 288 so
// static LDS stays < 64KB.
// ---------------------------------------------------------------------------
__global__ __launch_bounds__(128) void k_branch(
    const int* __restrict__ s, const int* __restrict__ sp, int mode,
    const float* __restrict__ w_norm,
    const float* __restrict__ w1, const float* __restrict__ b1,
    const __bf16* __restrict__ w2b, const float* __restrict__ b2,
    __bf16* __restrict__ y2)
{
    __shared__ float  wtab[64 * 8];                 // 2 KB
    __shared__ float  xpad[8 * 10 * 10];            // 3.2 KB (zero-padded input)
    __shared__ __bf16 bcol[64 * 288];               // 36 KB  im2col panel, col-major [px][k]

    int b    = blockIdx.x;
    int tid  = threadIdx.x;
    int lane = tid & 31;
    int wave = tid >> 5;
    int row  = lane & 15;
    int half = lane >> 4;

    for (int i = tid; i < 512; i += 128) wtab[i] = w_norm[i];
    for (int i = tid; i < 800; i += 128) xpad[i] = 0.f;
    __syncthreads();

    if (tid < 64) {
        int p = tid, ph = p >> 3, pw = p & 7;
        int is = s[b * 64 + p];
        int ip = sp[b * 64 + p];
        for (int c = 0; c < 8; ++c) {
            float es = wtab[is * 8 + c];
            float ep = wtab[ip * 8 + c];
            float v = (mode == 0) ? es : ((mode == 1) ? (ep - es) : ep);
            xpad[c * 100 + (ph + 1) * 10 + (pw + 1)] = v;
        }
    }
    __syncthreads();

    v8f acc[8];
#pragma unroll
    for (int mt = 0; mt < 8; ++mt)
        for (int j = 0; j < 8; ++j) acc[mt][j] = 0.f;

    for (int c0 = 0; c0 < 64; c0 += 32) {
        // zero im2col panel (covers implicit padding zeros)
        unsigned int* bz = (unsigned int*)bcol;
        for (int i = tid; i < (64 * 288) / 2; i += 128) bz[i] = 0u;
        __syncthreads();

        // conv1 for local channels [c0, c0+32), relu, scatter to bcol
        for (int o = tid; o < 32 * 64; o += 128) {
            int cl = o >> 6;               // local channel 0..31
            int p  = o & 63;
            int ph = p >> 3, pw = p & 7;
            int co1 = c0 + cl;
            const float* wrow = w1 + co1 * 72;
            float a1 = b1[co1];
            for (int ci = 0; ci < 8; ++ci) {
                const float* xc = xpad + ci * 100 + ph * 10 + pw;  // top-left of 3x3 window
                const float* wr = wrow + ci * 9;
                a1 += wr[0] * xc[0]  + wr[1] * xc[1]  + wr[2] * xc[2]
                    + wr[3] * xc[10] + wr[4] * xc[11] + wr[5] * xc[12]
                    + wr[6] * xc[20] + wr[7] * xc[21] + wr[8] * xc[22];
            }
            float v = a1 > 0.f ? a1 : 0.f;
            // y1[cl][ph][pw] contributes to output px (ph+1-kh, pw+1-kw)
            for (int kh = 0; kh < 3; ++kh) {
                int oh = ph + 1 - kh;
                if ((unsigned)oh >= 8u) continue;
                for (int kw = 0; kw < 3; ++kw) {
                    int ow = pw + 1 - kw;
                    if ((unsigned)ow >= 8u) continue;
                    int pp = oh * 8 + ow;
                    int k  = cl * 9 + kh * 3 + kw;
                    bcol[pp * 288 + k] = (__bf16)v;
                }
            }
        }
        __syncthreads();

        // WMMA: acc[mt] += W2[128 x 288_chunk] * bcol[288_chunk x 64]
        int kkbase = c0 * 9;
#pragma unroll
        for (int mt = 0; mt < 8; ++mt) {
            int co = mt * 16 + row;
            const __bf16* arow = w2b + co * K2_C + kkbase;
#pragma unroll
            for (int kt = 0; kt < 9; ++kt) {
                bf16x16 af, bf;
                const __bf16* ap = arow + kt * 32 + half * 8;
                ((uint4*)&af)[0] = *(const uint4*)(ap);
                ((uint4*)&af)[1] = *(const uint4*)(ap + 16);
                const __bf16* bp = bcol + (wave * 16 + row) * 288 + kt * 32 + half * 8;
                ((uint4*)&bf)[0] = *(const uint4*)(bp);
                ((uint4*)&bf)[1] = *(const uint4*)(bp + 16);
                acc[mt] = __builtin_amdgcn_wmma_f32_16x16x32_bf16(
                    false, af, false, bf, (short)0, acc[mt], false, false);
            }
        }
        __syncthreads();
    }

    // epilogue: bias + relu, store bf16 y2[b][co*64 + px]
    int px = wave * 16 + row;
#pragma unroll
    for (int mt = 0; mt < 8; ++mt) {
        for (int r = 0; r < 8; ++r) {
            int co = mt * 16 + r + half * 8;
            float v = acc[mt][r] + b2[co];
            v = v > 0.f ? v : 0.f;
            y2[(size_t)b * FEAT + co * 64 + px] = (__bf16)v;
        }
    }
}

// ---------------------------------------------------------------------------
// Register-blocked WMMA bf16 GEMM: C[M,N] = alpha * A[M,K] * Bt[N,K]^T + bias[N]
// A row-major, Bt row-major (i.e. B column-major).
// 128 threads = 4 waves; each wave computes a 32x64 tile (2 M-subtiles x
// 4 N-subtiles = 8 f32 accumulators), so each k-step issues 6 fragment
// loads for 8 WMMAs (0.75 loads/WMMA vs 2.0 for a single-tile wave),
// cutting L2 fragment traffic ~2.7x on the K=8192 linear layers.
// Block covers 128x64 of C; grid = (M/128, N/64). M%128==0, N%64==0 required.
// alpha = exp(scale[0]) if scale != null.
// ---------------------------------------------------------------------------
__global__ __launch_bounds__(128) void k_gemm(
    const __bf16* __restrict__ A, const __bf16* __restrict__ Bt,
    float* __restrict__ C, int M, int N, int K,
    const float* __restrict__ bias, const float* __restrict__ scaleptr)
{
    int lane = threadIdx.x & 31;
    int wave = threadIdx.x >> 5;
    int row  = lane & 15, half = lane >> 4;

    int m0 = blockIdx.x * 128 + wave * 32;   // wave's 32 rows
    int n0 = blockIdx.y * 64;                // block's 64 cols

    const __bf16* arow[2];
    arow[0] = A + (size_t)(m0 + row) * K;
    arow[1] = A + (size_t)(m0 + 16 + row) * K;
    const __bf16* brow[4];
#pragma unroll
    for (int nt = 0; nt < 4; ++nt)
        brow[nt] = Bt + (size_t)(n0 + nt * 16 + row) * K;

    v8f acc[2][4];
#pragma unroll
    for (int mi = 0; mi < 2; ++mi)
        for (int nt = 0; nt < 4; ++nt)
            for (int j = 0; j < 8; ++j) acc[mi][nt][j] = 0.f;

    for (int kt = 0; kt < K; kt += 32) {
        bf16x16 af[2], bf[4];
#pragma unroll
        for (int mi = 0; mi < 2; ++mi) {
            const __bf16* ap = arow[mi] + kt + half * 8;
            ((uint4*)&af[mi])[0] = *(const uint4*)(ap);
            ((uint4*)&af[mi])[1] = *(const uint4*)(ap + 16);
        }
#pragma unroll
        for (int nt = 0; nt < 4; ++nt) {
            const __bf16* bp = brow[nt] + kt + half * 8;
            ((uint4*)&bf[nt])[0] = *(const uint4*)(bp);
            ((uint4*)&bf[nt])[1] = *(const uint4*)(bp + 16);
        }
#pragma unroll
        for (int mi = 0; mi < 2; ++mi)
#pragma unroll
            for (int nt = 0; nt < 4; ++nt)
                acc[mi][nt] = __builtin_amdgcn_wmma_f32_16x16x32_bf16(
                    false, af[mi], false, bf[nt], (short)0, acc[mi][nt], false, false);
    }

    float alpha = scaleptr ? __expf(scaleptr[0]) : 1.0f;
#pragma unroll
    for (int nt = 0; nt < 4; ++nt) {
        int n = n0 + nt * 16 + row;
        float bv = bias ? bias[n] : 0.f;
#pragma unroll
        for (int mi = 0; mi < 2; ++mi) {
#pragma unroll
            for (int r = 0; r < 8; ++r) {
                int m = m0 + mi * 16 + r + half * 8;
                C[(size_t)m * N + n] = acc[mi][nt][r] * alpha + bv;
            }
        }
    }
}

// ---------------------------------------------------------------------------
// ||z||^2 per codebook row
// ---------------------------------------------------------------------------
__global__ void k_z2(const float* __restrict__ z, float* __restrict__ z2) {
    __shared__ float red[256];
    int n = blockIdx.x, t = threadIdx.x;
    float v = z[(size_t)n * 256 + t];
    red[t] = v * v;
    __syncthreads();
    for (int s = 128; s > 0; s >>= 1) {
        if (t < s) red[t] += red[t + s];
        __syncthreads();
    }
    if (t == 0) z2[n] = red[0];
}

// ---------------------------------------------------------------------------
// argmax_n of (z2[n] - 2*dot[b][n])  (== argmax of cdist; first index on tie)
// ---------------------------------------------------------------------------
__global__ void k_argmax(const float* __restrict__ dots, const float* __restrict__ z2,
                         int* __restrict__ inds) {
    __shared__ float sv[256];
    __shared__ int   si[256];
    int b = blockIdx.x, t = threadIdx.x;
    float best = -1e38f; int bi = 1 << 30;
    for (int n = t; n < NZ_C; n += 256) {
        float sc = z2[n] - 2.f * dots[(size_t)b * NZ_C + n];
        if (sc > best) { best = sc; bi = n; }
    }
    sv[t] = best; si[t] = bi;
    __syncthreads();
    for (int s = 128; s > 0; s >>= 1) {
        if (t < s) {
            if (sv[t + s] > sv[t] || (sv[t + s] == sv[t] && si[t + s] < si[t])) {
                sv[t] = sv[t + s]; si[t] = si[t + s];
            }
        }
        __syncthreads();
    }
    if (t == 0) inds[b] = si[0];
}

// ---------------------------------------------------------------------------
// out[b] = normalize(X[b] (+ z[inds[b]])) -> bf16   (256 threads per row)
// ---------------------------------------------------------------------------
__global__ void k_normalize(const float* __restrict__ X, const float* __restrict__ zvec,
                            const int* __restrict__ inds, __bf16* __restrict__ out) {
    __shared__ float red[256];
    int b = blockIdx.x, t = threadIdx.x;
    float v = X[(size_t)b * E_C + t];
    if (zvec) v += zvec[(size_t)inds[b] * E_C + t];
    red[t] = v * v;
    __syncthreads();
    for (int s = 128; s > 0; s >>= 1) {
        if (t < s) red[t] += red[t + s];
        __syncthreads();
    }
    float norm = sqrtf(red[0]);
    out[(size_t)b * E_C + t] = (__bf16)(v / (norm + 1e-4f));
}

// ---------------------------------------------------------------------------
// Host launcher
// ---------------------------------------------------------------------------
extern "C" void kernel_launch(void* const* d_in, const int* in_sizes, int n_in,
                              void* d_out, int out_size, void* d_ws, size_t ws_size,
                              hipStream_t stream) {
    const int*   s       = (const int*)d_in[0];
    const int*   sp      = (const int*)d_in[1];
    const float* w_embed = (const float*)d_in[2];
    const float* zv      = (const float*)d_in[21];
    const float* scale   = (const float*)d_in[22];

    char* ws = (char*)d_ws;
    size_t off = 0;
    auto carve = [&](size_t bytes) -> void* {
        void* p = ws + off;
        off += (bytes + 255) & ~(size_t)255;
        return p;
    };

    float*  w_norm = (float*)carve(64 * 8 * 4);
    __bf16* w2b    = (__bf16*)carve((size_t)C2_C * K2_C * 2);
    __bf16* lwb    = (__bf16*)carve((size_t)E_C * FEAT * 2);
    __bf16* y2     = (__bf16*)carve((size_t)B_SZ * FEAT * 2);
    float*  phi0   = (float*)carve((size_t)B_SZ * E_C * 4);   // s_int
    float*  phi1   = (float*)carve((size_t)B_SZ * E_C * 4);   // diff_int
    float*  phi2   = (float*)carve((size_t)B_SZ * E_C * 4);   // phi3 raw
    __bf16* zb     = (__bf16*)carve((size_t)NZ_C * E_C * 2);
    float*  z2     = (float*)carve(NZ_C * 4);
    __bf16* diffb  = (__bf16*)carve((size_t)B_SZ * E_C * 2);
    float*  dots   = (float*)carve((size_t)B_SZ * NZ_C * 4);
    int*    zinds  = (int*)carve(B_SZ * 4);
    __bf16* soutb  = (__bf16*)carve((size_t)B_SZ * E_C * 2);
    __bf16* spoutb = (__bf16*)carve((size_t)B_SZ * E_C * 2);
    float*  phis[3] = { phi0, phi1, phi2 };

    k_norm_embed<<<1, 64, 0, stream>>>(w_embed, w_norm);

    for (int br = 0; br < 3; ++br) {
        const float* w1 = (const float*)d_in[3 + br * 6 + 0];
        const float* b1 = (const float*)d_in[3 + br * 6 + 1];
        const float* w2 = (const float*)d_in[3 + br * 6 + 2];
        const float* b2 = (const float*)d_in[3 + br * 6 + 3];
        const float* lw = (const float*)d_in[3 + br * 6 + 4];
        const float* lb = (const float*)d_in[3 + br * 6 + 5];

        int nw2 = C2_C * K2_C;
        k_cvt_bf16<<<(nw2 + 255) / 256, 256, 0, stream>>>(w2, w2b, nw2);
        int nlw = E_C * FEAT;
        k_cvt_bf16<<<(nlw + 255) / 256, 256, 0, stream>>>(lw, lwb, nlw);

        k_branch<<<B_SZ, 128, 0, stream>>>(s, sp, br, w_norm, w1, b1, w2b, b2, y2);

        dim3 gl(B_SZ / 128, E_C / 64);
        k_gemm<<<gl, 128, 0, stream>>>(y2, lwb, phis[br], B_SZ, E_C, FEAT, lb, nullptr);
    }

    int nz = NZ_C * E_C;
    k_cvt_bf16<<<(nz + 255) / 256, 256, 0, stream>>>(zv, zb, nz);
    k_z2<<<NZ_C, 256, 0, stream>>>(zv, z2);
    int nd = B_SZ * E_C;
    k_cvt_bf16<<<(nd + 255) / 256, 256, 0, stream>>>(phi1, diffb, nd);

    dim3 gd(B_SZ / 128, NZ_C / 64);
    k_gemm<<<gd, 128, 0, stream>>>(diffb, zb, dots, B_SZ, NZ_C, E_C, nullptr, nullptr);

    k_argmax<<<B_SZ, 256, 0, stream>>>(dots, z2, zinds);

    k_normalize<<<B_SZ, 256, 0, stream>>>(phi0, zv, zinds, soutb);
    k_normalize<<<B_SZ, 256, 0, stream>>>(phi2, nullptr, nullptr, spoutb);

    dim3 gg(B_SZ / 128, B_SZ / 64);
    k_gemm<<<gg, 128, 0, stream>>>(soutb, spoutb, (float*)d_out, B_SZ, B_SZ, E_C, nullptr, scale);
}
